// NEModule_21784074125443
// MI455X (gfx1250) — matrix operand order repaired and checked by reading, not compiled, not run on hardware
//
#include <hip/hip_runtime.h>
#include <hip/hip_bf16.h>
#include <math.h>

// Point-Transformer block for MI455X (gfx1250, wave32, WMMA).
// Memory-bound (~200MB intermediate traffic @ 23.3 TB/s >> ~21.5 GFLOP of
// 128x128 GEMMs) -> f16 intermediates, f32 accumulation via
// v_wmma_f32_16x16x32_f16, all elementwise epilogues fused into the GEMMs,
// and an LDS-transposed epilogue so every HBM access is b128-coalesced.

#define BB 2
#define NN 4096
#define CC 128
#define KK 16
#define MM (BB * NN * KK)   // 131072 gathered points
#define PP (BB * NN)        // 8192 points
#define BN_EPS 1e-5f

typedef _Float16 f16;
typedef _Float16 v16h __attribute__((ext_vector_type(16)));
typedef _Float16 v8h  __attribute__((ext_vector_type(8)));
typedef float    v8f  __attribute__((ext_vector_type(8)));
typedef float    v4f  __attribute__((ext_vector_type(4)));

// ---------------------------------------------------------------- KNN
// One thread per query point; LDS-tiled scan of all N candidates keeping a
// sorted-ascending top-16 by squared distance (self included, d2=0 first,
// matching jax top_k on negative distances).
__global__ void knn_kernel(const float* __restrict__ xyz, int* __restrict__ idx) {
  const int p = blockIdx.x * blockDim.x + threadIdx.x;   // block never straddles b
  const int b = p / NN, n = p % NN;
  const float* xb = xyz + (size_t)b * 3 * NN;
  const float qx = xb[n], qy = xb[NN + n], qz = xb[2 * NN + n];
  float bd[KK]; int bi[KK];
#pragma unroll
  for (int t = 0; t < KK; ++t) { bd[t] = 3.4e38f; bi[t] = 0; }
  __shared__ float sx[256], sy[256], sz[256];
  for (int j0 = 0; j0 < NN; j0 += 256) {
    __syncthreads();
    const int j = j0 + threadIdx.x;
    sx[threadIdx.x] = xb[j];
    sy[threadIdx.x] = xb[NN + j];
    sz[threadIdx.x] = xb[2 * NN + j];
    __syncthreads();
    for (int jj = 0; jj < 256; ++jj) {
      const float dx = qx - sx[jj], dy = qy - sy[jj], dz = qz - sz[jj];
      const float d2 = dx * dx + dy * dy + dz * dz;
      if (d2 < bd[KK - 1]) {            // rare accept -> swap-chain insert
        float d = d2; int id = j0 + jj;
#pragma unroll
        for (int t = 0; t < KK; ++t) {
          if (d < bd[t]) { float td = bd[t]; int ti = bi[t]; bd[t] = d; bi[t] = id; d = td; id = ti; }
        }
      }
    }
  }
#pragma unroll
  for (int t = 0; t < KK; ++t) idx[(size_t)p * KK + t] = bi[t];
}

// ------------------------------------------------- layout/convert helpers
// features [B,C,N] f32 -> feat_h [B*N, C] f16 (row-major points, GEMM A layout)
__global__ void pack_feat(const float* __restrict__ feat, f16* __restrict__ feat_h) {
  const int p = blockIdx.x, c = threadIdx.x;
  const int b = p / NN, n = p % NN;
  feat_h[(size_t)p * CC + c] = (f16)feat[((size_t)b * CC + c) * NN + n];
}

__global__ void cvt_f32_f16(const float* __restrict__ src, f16* __restrict__ dst, int nElem) {
  const int i = blockIdx.x * blockDim.x + threadIdx.x;
  if (i < nElem) dst[i] = (f16)src[i];
}

// ------------------------------------------------- gather + theta1 (3->128)
// 256 threads handle 2 points m=(b,n,k); theta1 is 3 FMAs/output so it is
// computed inline instead of a GEMM. xj rows gathered from feat_h
// (contiguous per row after the pack transpose).
__global__ void gather_kernel(const float* __restrict__ xyz, const f16* __restrict__ feat_h,
                              const int* __restrict__ idx,
                              const float* __restrict__ w_t1, const float* __restrict__ b_t1,
                              f16* __restrict__ xj_h, f16* __restrict__ t1_h) {
  const int m = blockIdx.x * 2 + (threadIdx.x >> 7);
  const int c = threadIdx.x & 127;
  const int b = m / (NN * KK);
  const int n = (m / KK) % NN;
  const int j = idx[m];
  const float* xb = xyz + (size_t)b * 3 * NN;
  const float px = xb[n] - xb[j];
  const float py = xb[NN + n] - xb[NN + j];
  const float pz = xb[2 * NN + n] - xb[2 * NN + j];
  const float t = w_t1[c * 3 + 0] * px + w_t1[c * 3 + 1] * py + w_t1[c * 3 + 2] * pz + b_t1[c];
  t1_h[(size_t)m * CC + c] = (f16)t;
  xj_h[(size_t)m * CC + c] = feat_h[((size_t)b * NN + j) * CC + c];
}

// ------------------------------------------------- WMMA GEMM  O = A * W^T + b
// A [M,128] f16 row-major, W [128,128] f16 ([O,I] row-major, staged in LDS).
// Block = 256 threads = 8 waves; each wave owns a 16-row stripe and sweeps 8
// column tiles; K=128 via 4x v_wmma_f32_16x16x32_f16. The accumulator is
// bounced through a per-wave LDS staging tile (row stride 20 words -> no bank
// conflicts, b128-aligned) so the epilogue reads/writes HBM with contiguous
// b128 accesses instead of the 2B-granular D-fragment layout.
// Epilogue MODEs: 0 store f32 | 1 store f16 | 2 rel = linxi - v + delta (f16)
//                 | 3 feats = v + delta (f16)
#define STG_STRIDE 20
template <int MODE>
__global__ void gemm128(const f16* __restrict__ A, const f16* __restrict__ W,
                        const float* __restrict__ bias, void* __restrict__ out,
                        const float* __restrict__ linxi, const f16* __restrict__ delta) {
  __shared__ f16 lw[CC * CC] __attribute__((aligned(32)));
  __shared__ float stg[8][16 * STG_STRIDE] __attribute__((aligned(16)));
  const int tid = threadIdx.x;
  { // stage full 128x128 f16 weight (32KB) into LDS
    const uint32_t* ws = (const uint32_t*)W;
    uint32_t* wd = (uint32_t*)lw;
    for (int i = tid; i < CC * CC / 2; i += 256) wd[i] = ws[i];
  }
  __syncthreads();
  const int wave = tid >> 5;
  const int lane = tid & 31;
  const int row16 = lane & 15;      // col within tile for B/D, row for A
  const int khalf = lane >> 4;      // lane-half selects K/row sub-ranges
  const size_t rowBase = (size_t)blockIdx.x * 128 + (size_t)wave * 16;

  // A fragments (ISA 16-bit A 16x32 layout): lanes 0-15 hold K{0..7,16..23},
  // lanes 16-31 hold K{8..15,24..31} of row (lane&15).
  v16h afrag[4];
  const f16* arow = A + (rowBase + row16) * CC;
#pragma unroll
  for (int s = 0; s < 4; ++s) {
    v8h lo = *(const v8h*)(arow + s * 32 + khalf * 8);
    v8h hi = *(const v8h*)(arow + s * 32 + 16 + khalf * 8);
#pragma unroll
    for (int e = 0; e < 8; ++e) { afrag[s][e] = lo[e]; afrag[s][8 + e] = hi[e]; }
  }

  // transposed-epilogue lane mapping: each lane owns one half-row (8 cols)
  const int rowl = lane >> 1;        // 0..15 local output row
  const int halfc = lane & 1;        // which 8-col half
  const size_t grow = rowBase + rowl;

#pragma unroll 1
  for (int t = 0; t < 8; ++t) {
    const int col = t * 16 + row16;
    v8f acc = {};
#pragma unroll
    for (int s = 0; s < 4; ++s) {
      // B 32x16: lane = column, same K order as A: K{0..7,16..23} / K{8..15,24..31}.
      // B[k][col] = W[col][k] -> two 16B LDS reads per lane.
      v8h blo = *(const v8h*)(lw + col * CC + s * 32 + khalf * 8);
      v8h bhi = *(const v8h*)(lw + col * CC + s * 32 + 16 + khalf * 8);
      v16h bfrag;
#pragma unroll
      for (int e = 0; e < 8; ++e) { bfrag[e] = blo[e]; bfrag[8 + e] = bhi[e]; }
      acc = __builtin_amdgcn_wmma_f32_16x16x32_f16(false, afrag[s], false, bfrag,
                                                   (short)0, acc, false, false);
    }
    __syncthreads();   // previous tile's staging reads are complete
#pragma unroll
    for (int r = 0; r < 8; ++r)      // D layout -> LDS (banks conflict-free)
      stg[wave][(khalf * 8 + r) * STG_STRIDE + row16] = acc[r];
    __syncthreads();

    // read back one contiguous half-row per lane and run the epilogue on it
    const float* sp = &stg[wave][rowl * STG_STRIDE + halfc * 8];
    v4f a0 = *(const v4f*)(sp);
    v4f a1 = *(const v4f*)(sp + 4);
    float vv[8] = {a0[0], a0[1], a0[2], a0[3], a1[0], a1[1], a1[2], a1[3]};
    const int col0 = t * 16 + halfc * 8;
    const size_t o = grow * CC + col0;
    float bv[8];
#pragma unroll
    for (int e = 0; e < 8; ++e) bv[e] = bias[col0 + e];   // contiguous 32B

    if constexpr (MODE == 0) {
      float* op = (float*)out + o;
      v4f s0 = {vv[0] + bv[0], vv[1] + bv[1], vv[2] + bv[2], vv[3] + bv[3]};
      v4f s1 = {vv[4] + bv[4], vv[5] + bv[5], vv[6] + bv[6], vv[7] + bv[7]};
      *(v4f*)op = s0;
      *(v4f*)(op + 4) = s1;
    } else if constexpr (MODE == 1) {
      v8h hv;
#pragma unroll
      for (int e = 0; e < 8; ++e) hv[e] = (f16)(vv[e] + bv[e]);
      *(v8h*)((f16*)out + o) = hv;
    } else if constexpr (MODE == 2) {
      const v8h dv = *(const v8h*)(delta + o);                   // contiguous b128
      const float* lx = linxi + (grow >> 4) * CC + col0;         // contiguous 32B
      v8h hv;
#pragma unroll
      for (int e = 0; e < 8; ++e)
        hv[e] = (f16)(lx[e] - (vv[e] + bv[e]) + (float)dv[e]);
      *(v8h*)((f16*)out + o) = hv;
    } else {
      const v8h dv = *(const v8h*)(delta + o);
      v8h hv;
#pragma unroll
      for (int e = 0; e < 8; ++e) hv[e] = (f16)(vv[e] + bv[e] + (float)dv[e]);
      *(v8h*)((f16*)out + o) = hv;
    }
  }
}

// ------------------------------------------------- BatchNorm (training stats)
// Pass 1: per-channel sum / sumsq over all M rows (block partials + f32 atomics)
__global__ void colstats(const f16* __restrict__ X, float* __restrict__ sums, int Mrows) {
  __shared__ float ss[256], sq[256];
  const int tid = threadIdx.x;
  const int c = tid & 127;
  const int half = tid >> 7;
  const int rowsPer = Mrows / gridDim.x;
  const int r0 = blockIdx.x * rowsPer;
  float s = 0.f, q = 0.f;
  for (int r = r0 + half; r < r0 + rowsPer; r += 2) {
    const float v = (float)X[(size_t)r * CC + c];
    s += v; q += v * v;
  }
  ss[tid] = s; sq[tid] = q;
  __syncthreads();
  if (half == 0) {
    atomicAdd(&sums[c], ss[c] + ss[c + 128]);
    atomicAdd(&sums[128 + c], sq[c] + sq[c + 128]);
  }
}

// Pass 2: in-place y = relu(g*(x-mean)*rsqrt(var+eps)+beta)
__global__ void bn_relu(f16* __restrict__ X, const float* __restrict__ sums,
                        const float* __restrict__ g, const float* __restrict__ be, int Mrows) {
  const size_t i = (size_t)blockIdx.x * blockDim.x + threadIdx.x;
  const int c = (int)(i & 127);
  const float invM = 1.0f / (float)Mrows;
  const float mean = sums[c] * invM;
  const float var = sums[128 + c] * invM - mean * mean;
  const float sc = g[c] * rsqrtf(var + BN_EPS);
  float v = ((float)X[i] - mean) * sc + be[c];
  X[i] = (f16)(v > 0.f ? v : 0.f);
}

// ------------------------------------------------- softmax over K + reduce
__global__ void softmax_out(const f16* __restrict__ rel2, const f16* __restrict__ feats,
                            float* __restrict__ out) {
  const int p = blockIdx.x, c = threadIdx.x;
  const int b = p / NN, n = p % NN;
  const size_t base = (size_t)p * KK * CC + c;
  float w[KK];
  float mx = -3.4e38f;
#pragma unroll
  for (int k = 0; k < KK; ++k) { w[k] = (float)rel2[base + (size_t)k * CC]; mx = fmaxf(mx, w[k]); }
  float sum = 0.f;
#pragma unroll
  for (int k = 0; k < KK; ++k) { w[k] = __expf(w[k] - mx); sum += w[k]; }
  const float inv = 1.f / sum;
  float o = 0.f;
#pragma unroll
  for (int k = 0; k < KK; ++k) o += w[k] * inv * (float)feats[base + (size_t)k * CC];
  out[((size_t)b * CC + c) * NN + n] = o;
}

// ---------------------------------------------------------------- launch
extern "C" void kernel_launch(void* const* d_in, const int* in_sizes, int n_in,
                              void* d_out, int out_size, void* d_ws, size_t ws_size,
                              hipStream_t stream) {
  const float* xyz      = (const float*)d_in[0];
  const float* features = (const float*)d_in[1];
  const float* w_theta1 = (const float*)d_in[2];
  const float* b_theta1 = (const float*)d_in[3];
  const float* w_theta2 = (const float*)d_in[4];
  const float* b_theta2 = (const float*)d_in[5];
  const float* g_theta  = (const float*)d_in[6];
  const float* be_theta = (const float*)d_in[7];
  const float* w_phi    = (const float*)d_in[8];
  const float* b_phi    = (const float*)d_in[9];
  const float* w_psi    = (const float*)d_in[10];
  const float* b_psi    = (const float*)d_in[11];
  const float* w_alpha  = (const float*)d_in[12];
  const float* b_alpha  = (const float*)d_in[13];
  const float* w_gamma1 = (const float*)d_in[14];
  const float* b_gamma1 = (const float*)d_in[15];
  const float* w_gamma2 = (const float*)d_in[16];
  const float* b_gamma2 = (const float*)d_in[17];
  const float* g_gamma  = (const float*)d_in[18];
  const float* be_gamma = (const float*)d_in[19];
  float* out = (float*)d_out;

  // workspace carve-up (256B aligned slices)
  size_t off = 0;
  auto carve = [&](size_t bytes) {
    void* p = (char*)d_ws + off;
    off += (bytes + 255) & ~(size_t)255;
    return p;
  };
  int*   idx     = (int*)  carve((size_t)MM * 4);
  f16*   feat_h  = (f16*)  carve((size_t)PP * CC * 2);
  f16*   xj_h    = (f16*)  carve((size_t)MM * CC * 2);
  f16*   t1_h    = (f16*)  carve((size_t)MM * CC * 2);   // reused as gamma1 out
  f16*   dpre_h  = (f16*)  carve((size_t)MM * CC * 2);   // theta2 out -> delta
  f16*   rel_h   = (f16*)  carve((size_t)MM * CC * 2);
  f16*   g2_h    = (f16*)  carve((size_t)MM * CC * 2);   // gamma2 out -> rel2
  f16*   feats_h = (f16*)  carve((size_t)MM * CC * 2);
  float* linxi   = (float*)carve((size_t)PP * CC * 4);
  f16*   wh      = (f16*)  carve((size_t)6 * CC * CC * 2);
  float* stats   = (float*)carve((size_t)2 * 256 * 4);   // [set][sum|sumsq][128]

  f16* wh_theta2 = wh + 0 * CC * CC;
  f16* wh_phi    = wh + 1 * CC * CC;
  f16* wh_psi    = wh + 2 * CC * CC;
  f16* wh_g1     = wh + 3 * CC * CC;
  f16* wh_g2     = wh + 4 * CC * CC;
  f16* wh_alpha  = wh + 5 * CC * CC;
  float* stats0 = stats;        // theta BN
  float* stats1 = stats + 256;  // gamma BN

  // 1) KNN + layout packs
  knn_kernel<<<PP / 256, 256, 0, stream>>>(xyz, idx);
  pack_feat<<<PP, CC, 0, stream>>>(features, feat_h);
  const int WN = CC * CC;
  cvt_f32_f16<<<WN / 256, 256, 0, stream>>>(w_theta2, wh_theta2, WN);
  cvt_f32_f16<<<WN / 256, 256, 0, stream>>>(w_phi,    wh_phi,    WN);
  cvt_f32_f16<<<WN / 256, 256, 0, stream>>>(w_psi,    wh_psi,    WN);
  cvt_f32_f16<<<WN / 256, 256, 0, stream>>>(w_gamma1, wh_g1,     WN);
  cvt_f32_f16<<<WN / 256, 256, 0, stream>>>(w_gamma2, wh_g2,     WN);
  cvt_f32_f16<<<WN / 256, 256, 0, stream>>>(w_alpha,  wh_alpha,  WN);
  hipMemsetAsync(stats, 0, 2 * 256 * 4, stream);

  // 2) gather x_j + theta1(pos)
  gather_kernel<<<MM / 2, 256, 0, stream>>>(xyz, feat_h, idx, w_theta1, b_theta1, xj_h, t1_h);

  // 3) delta = relu(BN(theta2(t1)))
  gemm128<1><<<MM / 128, 256, 0, stream>>>(t1_h, wh_theta2, b_theta2, dpre_h, nullptr, nullptr);
  colstats<<<512, 256, 0, stream>>>(dpre_h, stats0, MM);
  bn_relu<<<(size_t)MM * CC / 256, 256, 0, stream>>>(dpre_h, stats0, g_theta, be_theta, MM);

  // 4) rel = phi(x_i) - psi(x_j) + delta   (psi epilogue fuses the combine)
  gemm128<0><<<PP / 128, 256, 0, stream>>>(feat_h, wh_phi, b_phi, linxi, nullptr, nullptr);
  gemm128<2><<<MM / 128, 256, 0, stream>>>(xj_h, wh_psi, b_psi, rel_h, linxi, dpre_h);

  // 5) rel2 = relu(BN(gamma2(gamma1(rel))))
  gemm128<1><<<MM / 128, 256, 0, stream>>>(rel_h, wh_g1, b_gamma1, t1_h, nullptr, nullptr);
  gemm128<1><<<MM / 128, 256, 0, stream>>>(t1_h, wh_g2, b_gamma2, g2_h, nullptr, nullptr);
  colstats<<<512, 256, 0, stream>>>(g2_h, stats1, MM);
  bn_relu<<<(size_t)MM * CC / 256, 256, 0, stream>>>(g2_h, stats1, g_gamma, be_gamma, MM);

  // 6) feats = alpha(x_j) + delta (fused epilogue)
  gemm128<3><<<MM / 128, 256, 0, stream>>>(xj_h, wh_alpha, b_alpha, feats_h, nullptr, dpre_h);

  // 7) out = sum_k softmax_k(rel2) * feats
  softmax_out<<<PP, CC, 0, stream>>>(g2_h, feats_h, out);

  (void)in_sizes; (void)n_in; (void)out_size; (void)ws_size;
}
